// HierarchicalCausalTransformer_32435593019962
// MI455X (gfx1250) — matrix operand order repaired and checked by reading, not compiled, not run on hardware
//
#include <hip/hip_runtime.h>
#include <hip/hip_bf16.h>
#include <math.h>

// ---------------------------------------------------------------------------
// MI455X (gfx1250) implementation: all GEMMs via v_wmma_f32_16x16x32_bf16,
// wave32, 8 waves/block, 128x64 block tile, K-step 32.
// Staging: float4 global loads -> bf16 LDS; fragments via b128 LDS loads.
// ---------------------------------------------------------------------------

typedef __attribute__((ext_vector_type(16))) __bf16 v16bf;
typedef __attribute__((ext_vector_type(8)))  float  v8f;

constexpr int DMODEL = 512;
constexpr int NHEADS = 8;
constexpr int DHEAD  = 64;
constexpr int VOCAB  = 8192;
constexpr int SEQ0   = 1024;
constexpr int SUB    = 4;
constexpr int BATCH  = 2;
constexpr int TOK0   = BATCH * SEQ0;        // 2048 stage-0 tokens
constexpr int TOK1   = BATCH * SEQ0 * SUB;  // 8192 stage-1 tokens

constexpr int TBM = 128;   // block tile M
constexpr int TBN = 64;    // block tile N
constexpr int TKT = 32;    // K step (= bf16 WMMA K)
constexpr int TKA = 40;    // LDS row stride in bf16 (80B: 16B-aligned, bank-spread)

union FragBF { v16bf v; uint4 q[2]; };
union PackBF4 { __bf16 h[4]; uint2 u2; };

// ---------------------------------------------------------------------------
// Generic batched GEMM:  C = act(alpha * A*B + bias) + resid
// A: fp32, [M x K], row stride lda   (converted to bf16 at LDS stage)
// B: fp32, [K x N] row-major (ldb = row stride) or transB: [N x K] row-major
// Two-level batch: z -> (zo = z/zdiv, zi = z%zdiv), pointer offsets per level.
// ---------------------------------------------------------------------------
__global__ __launch_bounds__(256)
void gemm_bf16_wmma(const float* __restrict__ A, const float* __restrict__ B,
                    float* __restrict__ C, const float* __restrict__ bias,
                    const float* __restrict__ resid,
                    int M, int N, int K, int lda, int ldb, int ldc,
                    long long sAo, long long sAi, long long sBo, long long sBi,
                    long long sCo, long long sCi,
                    int zdiv, int transB, int actGelu, float alpha)
{
    __shared__ alignas(16) __bf16 As[TBM][TKA];   // A tile [m][k]
    __shared__ alignas(16) __bf16 Bt[TBN][TKA];   // B tile transposed: [n][k]

    int z  = blockIdx.z;
    int zo = z / zdiv, zi = z % zdiv;
    A += zo * sAo + zi * sAi;
    B += zo * sBo + zi * sBi;
    C += zo * sCo + zi * sCi;
    if (resid) resid += zo * sCo + zi * sCi;

    const int tid  = threadIdx.x;
    const int lane = tid & 31;
    const int wave = tid >> 5;
    const int wm   = wave & 3;   // wave M position (0..3)
    const int wn   = wave >> 2;  // wave N position (0..1)
    const int m0   = blockIdx.y * TBM;
    const int n0   = blockIdx.x * TBN;

    const int mr  = lane & 15;   // row/col within 16-wide fragment
    const int sel = lane >> 4;   // half-wave select (ISA 7.12.2 layouts)

    const bool fullMN = (m0 + TBM <= M) && (n0 + TBN <= N);

    v8f acc[2][2] = {};

    for (int k0 = 0; k0 < K; k0 += TKT) {
        if (fullMN && (k0 + TKT <= K)) {
            // ================= fast path: vector loads, no guards ==========
            // A tile: 128x32 floats = 1024 float4, 4 per thread
            #pragma unroll
            for (int i = 0; i < 4; ++i) {
                int idx = tid + i * 256;          // 0..1023
                int r   = idx >> 3;               // 8 float4 per row
                int c4  = (idx & 7) * 4;
                const float4 f = *(const float4*)&A[(long long)(m0 + r) * lda + k0 + c4];
                PackBF4 p;
                p.h[0] = (__bf16)f.x; p.h[1] = (__bf16)f.y;
                p.h[2] = (__bf16)f.z; p.h[3] = (__bf16)f.w;
                *(uint2*)&As[r][c4] = p.u2;
            }
            // B tile: 32x64 floats = 512 float4, 2 per thread
            if (transB) {
                // B stored [N][K]: contiguous along k -> vector load + b64 LDS store
                #pragma unroll
                for (int i = 0; i < 2; ++i) {
                    int idx = tid + i * 256;      // 0..511
                    int n   = idx >> 3;           // 8 float4 per n-row
                    int c4  = (idx & 7) * 4;
                    const float4 f = *(const float4*)&B[(long long)(n0 + n) * ldb + k0 + c4];
                    PackBF4 p;
                    p.h[0] = (__bf16)f.x; p.h[1] = (__bf16)f.y;
                    p.h[2] = (__bf16)f.z; p.h[3] = (__bf16)f.w;
                    *(uint2*)&Bt[n][c4] = p.u2;
                }
            } else {
                // B stored [K][N]: contiguous along n -> vector load, transpose store
                #pragma unroll
                for (int i = 0; i < 2; ++i) {
                    int idx = tid + i * 256;      // 0..511
                    int k   = idx >> 4;           // 16 float4 per k-row
                    int c4  = (idx & 15) * 4;
                    const float4 f = *(const float4*)&B[(long long)(k0 + k) * ldb + n0 + c4];
                    Bt[c4 + 0][k] = (__bf16)f.x;
                    Bt[c4 + 1][k] = (__bf16)f.y;
                    Bt[c4 + 2][k] = (__bf16)f.z;
                    Bt[c4 + 3][k] = (__bf16)f.w;
                }
            }
        } else {
            // ================= edge path: branch-free clamp + select =======
            #pragma unroll
            for (int i = 0; i < (TBM * TKT) / 256; ++i) {
                int idx = tid + i * 256;
                int r = idx / TKT, c = idx % TKT;
                int gm = m0 + r, gk = k0 + c;
                int cm = gm < M - 1 ? gm : M - 1;
                int ck = gk < K - 1 ? gk : K - 1;
                float v = A[(long long)cm * lda + ck];
                v = (gm < M && gk < K) ? v : 0.0f;
                As[r][c] = (__bf16)v;
            }
            #pragma unroll
            for (int i = 0; i < (TKT * TBN) / 256; ++i) {
                int idx = tid + i * 256;
                int r = idx / TBN, c = idx % TBN;   // r = k, c = n
                int gk = k0 + r, gn = n0 + c;
                int ck = gk < K - 1 ? gk : K - 1;
                int cn = gn < N - 1 ? gn : N - 1;
                float v = transB ? B[(long long)cn * ldb + ck]
                                 : B[(long long)ck * ldb + cn];
                v = (gk < K && gn < N) ? v : 0.0f;
                Bt[c][r] = (__bf16)v;
            }
        }
        __syncthreads();

        // CDNA5 prefetch of next K tile (-> global_prefetch_b8)
        if (k0 + TKT < K && (m0 + (tid & 127)) < M)
            __builtin_prefetch(&A[(long long)(m0 + (tid & 127)) * lda + k0 + TKT], 0, 1);

        // ---- fragments as contiguous b128 LDS loads (ISA 7.12.2 layouts) ----
        FragBF afr[2], bfr[2];
        #pragma unroll
        for (int i = 0; i < 2; ++i) {
            int rb = wm * 32 + i * 16 + mr;
            afr[i].q[0] = *(const uint4*)&As[rb][sel * 8];
            afr[i].q[1] = *(const uint4*)&As[rb][16 + sel * 8];
        }
        #pragma unroll
        for (int j = 0; j < 2; ++j) {
            int cb = wn * 32 + j * 16 + mr;
            bfr[j].q[0] = *(const uint4*)&Bt[cb][sel * 8];
            bfr[j].q[1] = *(const uint4*)&Bt[cb][16 + sel * 8];
        }

        #pragma unroll
        for (int i = 0; i < 2; ++i)
            #pragma unroll
            for (int j = 0; j < 2; ++j)
                acc[i][j] = __builtin_amdgcn_wmma_f32_16x16x32_bf16(
                    false, afr[i].v, false, bfr[j].v, (short)0, acc[i][j], false, false);

        __syncthreads();
    }

    // ---- epilogue per 32-bit C/D layout: VGPR v -> M = v + 8*sel, lane -> N ----
    #pragma unroll
    for (int i = 0; i < 2; ++i) {
        #pragma unroll
        for (int j = 0; j < 2; ++j) {
            #pragma unroll
            for (int v = 0; v < 8; ++v) {
                int gm = m0 + wm * 32 + i * 16 + v + 8 * sel;
                int gn = n0 + wn * 32 + j * 16 + mr;
                if (gm < M && gn < N) {
                    float val = acc[i][j][v] * alpha;
                    if (bias)    val += bias[gn];
                    if (actGelu) val = 0.5f * val * (1.0f + erff(val * 0.70710678118654752f));
                    if (resid)   val += resid[(long long)gm * ldc + gn];
                    C[(long long)gm * ldc + gn] = val;
                }
            }
        }
    }
}

// ---------------------------------------------------------------------------
// LayerNorm over D=512, one block (256 threads) per row.
// ---------------------------------------------------------------------------
__global__ __launch_bounds__(256)
void layernorm_k(const float* __restrict__ x, const float* __restrict__ w,
                 const float* __restrict__ b, float* __restrict__ y, int rows)
{
    int row = blockIdx.x;
    if (row >= rows) return;
    const float* xr = x + (long long)row * DMODEL;
    float* yr       = y + (long long)row * DMODEL;
    int tid = threadIdx.x;
    float v0 = xr[tid], v1 = xr[tid + 256];
    float s = v0 + v1, sq = v0 * v0 + v1 * v1;
    #pragma unroll
    for (int o = 16; o > 0; o >>= 1) {
        s  += __shfl_xor(s,  o, 32);
        sq += __shfl_xor(sq, o, 32);
    }
    __shared__ float rs[8], rq[8];
    __shared__ float mean_s, rstd_s;
    if ((tid & 31) == 0) { rs[tid >> 5] = s; rq[tid >> 5] = sq; }
    __syncthreads();
    if (tid == 0) {
        float ts = 0.f, tq = 0.f;
        #pragma unroll
        for (int i = 0; i < 8; ++i) { ts += rs[i]; tq += rq[i]; }
        float m   = ts / (float)DMODEL;
        float var = tq / (float)DMODEL - m * m;
        mean_s = m;
        rstd_s = rsqrtf(var + 1e-5f);
    }
    __syncthreads();
    float m = mean_s, r = rstd_s;
    yr[tid]       = (v0 - m) * r * w[tid]       + b[tid];
    yr[tid + 256] = (v1 - m) * r * w[tid + 256] + b[tid + 256];
}

// ---------------------------------------------------------------------------
// Causal softmax with ALiBi-style bias (bias = j * slope_h, slopes 2^-(h+1)).
// In-place on S[z, row, 0..n). grid = (n, nbatch*H)
// ---------------------------------------------------------------------------
__global__ __launch_bounds__(256)
void softmax_alibi_k(float* __restrict__ S, int n)
{
    int row = blockIdx.x;
    int z   = blockIdx.y;
    int h   = z % NHEADS;
    float slope = exp2f(-(float)(h + 1));
    float* Sr = S + ((long long)z * n + row) * (long long)n;
    int tid = threadIdx.x;
    const float NEG = -3.402823466e38f;

    float xv[4];                       // n <= 1024, 256 threads -> <=4 each
    int cnt = 0;
    float mx = NEG;
    for (int j = tid; j < n; j += 256) {
        float x = (j <= row) ? (Sr[j] + slope * (float)j) : NEG;
        xv[cnt++] = x;
        mx = fmaxf(mx, x);
    }
    #pragma unroll
    for (int o = 16; o > 0; o >>= 1) mx = fmaxf(mx, __shfl_xor(mx, o, 32));
    __shared__ float red[8];
    __shared__ float mx_s, sum_s;
    if ((tid & 31) == 0) red[tid >> 5] = mx;
    __syncthreads();
    if (tid == 0) {
        float m = red[0];
        #pragma unroll
        for (int i = 1; i < 8; ++i) m = fmaxf(m, red[i]);
        mx_s = m;
    }
    __syncthreads();
    mx = mx_s;

    float sum = 0.f;
    cnt = 0;
    for (int j = tid; j < n; j += 256) {
        float e = __expf(xv[cnt] - mx);
        xv[cnt++] = e;
        sum += e;
    }
    #pragma unroll
    for (int o = 16; o > 0; o >>= 1) sum += __shfl_xor(sum, o, 32);
    __syncthreads();
    if ((tid & 31) == 0) red[tid >> 5] = sum;
    __syncthreads();
    if (tid == 0) {
        float t = 0.f;
        #pragma unroll
        for (int i = 0; i < 8; ++i) t += red[i];
        sum_s = t;
    }
    __syncthreads();
    float inv = 1.0f / sum_s;
    cnt = 0;
    for (int j = tid; j < n; j += 256) Sr[j] = xv[cnt++] * inv;
}

// ---------------------------------------------------------------------------
// Input builders
// ---------------------------------------------------------------------------
// x0[b,t,:] = start (t==0) else sum_j token_emb[ids[b,t-1,j]] + pos_emb0[t-1]
__global__ void build_x0_k(const int* __restrict__ ids, const float* __restrict__ tok_emb,
                           const float* __restrict__ start, const float* __restrict__ pos0,
                           float* __restrict__ x0)
{
    long long idx = (long long)blockIdx.x * blockDim.x + threadIdx.x;
    if (idx >= (long long)TOK0 * DMODEL) return;
    int d   = (int)(idx % DMODEL);
    int row = (int)(idx / DMODEL);           // b*1024 + t
    int t   = row % SEQ0;
    float val;
    if (t == 0) {
        val = start[d];
    } else {
        int base = (row - 1) * SUB;          // (b*1024 + t-1) * 4
        float s = 0.f;
        #pragma unroll
        for (int j = 0; j < SUB; ++j)
            s += tok_emb[(long long)ids[base + j] * DMODEL + d];
        val = s + pos0[(long long)(t - 1) * DMODEL + d];
    }
    x0[idx] = val;
}

// stage1_in[ti,:] = token_emb[ids[ti]] + pos_emb1[ti % 4]
__global__ void embed_s1in_k(const int* __restrict__ ids, const float* __restrict__ tok_emb,
                             const float* __restrict__ pos1, float* __restrict__ s1in)
{
    long long idx = (long long)blockIdx.x * blockDim.x + threadIdx.x;
    if (idx >= (long long)TOK1 * DMODEL) return;
    int d = (int)(idx % DMODEL);
    long long ti = idx / DMODEL;
    int l1 = (int)(ti % SUB);
    s1in[idx] = tok_emb[(long long)ids[ti] * DMODEL + d] + pos1[(long long)l1 * DMODEL + d];
}

// x1[g, j, :] = a0[g,:] (j==0) else stage1_in[g, j-1, :]
__global__ void build_x1_k(const float* __restrict__ a0, const float* __restrict__ s1in,
                           float* __restrict__ x1)
{
    long long idx = (long long)blockIdx.x * blockDim.x + threadIdx.x;
    if (idx >= (long long)TOK1 * DMODEL) return;
    int d = (int)(idx % DMODEL);
    long long ti = idx / DMODEL;             // (b*1024+l0)*4 + j
    int j = (int)(ti % SUB);
    long long g = ti / SUB;                  // b*1024 + l0
    x1[idx] = (j == 0) ? a0[g * DMODEL + d]
                       : s1in[(g * SUB + (j - 1)) * DMODEL + d];
}

// ---------------------------------------------------------------------------
// Host orchestration
// ---------------------------------------------------------------------------
extern "C" void kernel_launch(void* const* d_in, const int* in_sizes, int n_in,
                              void* d_out, int out_size, void* d_ws, size_t ws_size,
                              hipStream_t stream)
{
    (void)in_sizes; (void)n_in; (void)out_size; (void)ws_size;

    const int*   ids     = (const int*)d_in[0];
    const float* tok_emb = (const float*)d_in[1];
    const float* startv  = (const float*)d_in[2];
    const float* pos0    = (const float*)d_in[3];
    const float* pos1    = (const float*)d_in[4];
    const int T0_BASE = 5;                 // 8 layers x 10 tensors
    const int T0_NW = 85, T0_NB = 86;
    const int T1_BASE = 87;                // 4 layers x 10 tensors
    const int T1_NW = 127, T1_NB = 128;
    const int W_LOGITS = 129, B_LOGITS = 130;

    // workspace partition (fp32)
    float* wsp = (float*)d_ws;
    float* X    = wsp; wsp += (long long)TOK1 * DMODEL;      // activations (both stages)
    float* Hb   = wsp; wsp += (long long)TOK1 * DMODEL;      // normed hidden
    float* QKV  = wsp; wsp += (long long)TOK1 * 3 * DMODEL;  // qkv
    float* ATT  = wsp; wsp += (long long)TOK1 * DMODEL;      // attention output
    float* FF   = wsp; wsp += (long long)TOK1 * 4 * DMODEL;  // FF hidden
    float* S    = wsp; wsp += (long long)BATCH * NHEADS * SEQ0 * SEQ0; // scores (stage0 max)
    float* S1IN = wsp; wsp += (long long)TOK1 * DMODEL;      // tok + pos_emb1
    float* A0   = wsp; wsp += (long long)TOK0 * DMODEL;      // stage0 final LN output

    auto gemm = [&](const float* A, const float* B, float* C,
                    const float* bias, const float* resid,
                    int M, int N, int K, int lda, int ldb, int ldc,
                    long long sAo, long long sAi, long long sBo, long long sBi,
                    long long sCo, long long sCi,
                    int nbatch, int zdiv, int transB, int actGelu, float alpha)
    {
        dim3 grid((N + TBN - 1) / TBN, (M + TBM - 1) / TBM, nbatch);
        gemm_bf16_wmma<<<grid, dim3(256), 0, stream>>>(
            A, B, C, bias, resid, M, N, K, lda, ldb, ldc,
            sAo, sAi, sBo, sBi, sCo, sCi, zdiv, transB, actGelu, alpha);
    };

    auto run_stage = [&](float* Xb, int T, int n, int nb, int base, int nlayers)
    {
        for (int li = 0; li < nlayers; ++li) {
            const float* an_w = (const float*)d_in[base + li * 10 + 0];
            const float* an_b = (const float*)d_in[base + li * 10 + 1];
            const float* wqkv = (const float*)d_in[base + li * 10 + 2];
            const float* wo   = (const float*)d_in[base + li * 10 + 3];
            const float* fn_w = (const float*)d_in[base + li * 10 + 4];
            const float* fn_b = (const float*)d_in[base + li * 10 + 5];
            const float* w1   = (const float*)d_in[base + li * 10 + 6];
            const float* b1   = (const float*)d_in[base + li * 10 + 7];
            const float* w2   = (const float*)d_in[base + li * 10 + 8];
            const float* b2   = (const float*)d_in[base + li * 10 + 9];

            // h = LN(x)
            layernorm_k<<<dim3(T), dim3(256), 0, stream>>>(Xb, an_w, an_b, Hb, T);
            // qkv = h @ wqkv
            gemm(Hb, wqkv, QKV, nullptr, nullptr, T, 3 * DMODEL, DMODEL,
                 DMODEL, 3 * DMODEL, 3 * DMODEL,
                 0, 0, 0, 0, 0, 0,
                 1, 1, 0, 0, 1.0f);
            // S = (1/sqrt(dh)) * Q @ K^T    batched over (seq-batch, head)
            gemm(QKV + 0, QKV + DMODEL, S, nullptr, nullptr,
                 n, n, DHEAD, 3 * DMODEL, 3 * DMODEL, n,
                 (long long)n * 3 * DMODEL, DHEAD,
                 (long long)n * 3 * DMODEL, DHEAD,
                 (long long)NHEADS * n * n, (long long)n * n,
                 nb * NHEADS, NHEADS, /*transB=*/1, 0, 0.125f);
            // softmax with ALiBi bias + causal mask
            softmax_alibi_k<<<dim3(n, nb * NHEADS), dim3(256), 0, stream>>>(S, n);
            // O = P @ V  -> ATT (heads concatenated)
            gemm(S, QKV + 2 * DMODEL, ATT, nullptr, nullptr,
                 n, DHEAD, n, n, 3 * DMODEL, DMODEL,
                 (long long)NHEADS * n * n, (long long)n * n,
                 (long long)n * 3 * DMODEL, DHEAD,
                 (long long)n * DMODEL, DHEAD,
                 nb * NHEADS, NHEADS, 0, 0, 1.0f);
            // x = O @ wo + x
            gemm(ATT, wo, Xb, nullptr, Xb, T, DMODEL, DMODEL,
                 DMODEL, DMODEL, DMODEL,
                 0, 0, 0, 0, 0, 0,
                 1, 1, 0, 0, 1.0f);
            // h = LN(x); ff = gelu(h @ w1 + b1); x = ff @ w2 + b2 + x
            layernorm_k<<<dim3(T), dim3(256), 0, stream>>>(Xb, fn_w, fn_b, Hb, T);
            gemm(Hb, w1, FF, b1, nullptr, T, 4 * DMODEL, DMODEL,
                 DMODEL, 4 * DMODEL, 4 * DMODEL,
                 0, 0, 0, 0, 0, 0,
                 1, 1, /*transB=*/0, /*gelu=*/1, 1.0f);
            gemm(FF, w2, Xb, b2, Xb, T, DMODEL, 4 * DMODEL,
                 4 * DMODEL, DMODEL, DMODEL,
                 0, 0, 0, 0, 0, 0,
                 1, 1, 0, 0, 1.0f);
        }
    };

    // ---- build inputs ----
    long long tot0 = (long long)TOK0 * DMODEL;
    build_x0_k<<<dim3((unsigned)((tot0 + 255) / 256)), dim3(256), 0, stream>>>(
        ids, tok_emb, startv, pos0, X);
    long long tot1 = (long long)TOK1 * DMODEL;
    embed_s1in_k<<<dim3((unsigned)((tot1 + 255) / 256)), dim3(256), 0, stream>>>(
        ids, tok_emb, pos1, S1IN);

    // ---- stage 0: 8 layers, n=1024, 2 sequences ----
    run_stage(X, TOK0, SEQ0, BATCH, T0_BASE, 8);
    layernorm_k<<<dim3(TOK0), dim3(256), 0, stream>>>(
        X, (const float*)d_in[T0_NW], (const float*)d_in[T0_NB], A0, TOK0);

    // ---- assemble stage-1 input ----
    build_x1_k<<<dim3((unsigned)((tot1 + 255) / 256)), dim3(256), 0, stream>>>(A0, S1IN, X);

    // ---- stage 1: 4 layers, n=4, 2048 sequences ----
    run_stage(X, TOK1, SUB, BATCH * SEQ0, T1_BASE, 4);
    layernorm_k<<<dim3(TOK1), dim3(256), 0, stream>>>(
        X, (const float*)d_in[T1_NW], (const float*)d_in[T1_NB], Hb, TOK1);

    // ---- logits: (8192 x 512) @ (512 x 8192) + b ----
    gemm(Hb, (const float*)d_in[W_LOGITS], (float*)d_out,
         (const float*)d_in[B_LOGITS], nullptr,
         TOK1, VOCAB, DMODEL, DMODEL, VOCAB, VOCAB,
         0, 0, 0, 0, 0, 0,
         1, 1, 0, 0, 1.0f);
}
